// SlidingWindowAttention_36859409334433
// MI455X (gfx1250) — matrix operand order repaired and checked by reading, not compiled, not run on hardware
//
#include <hip/hip_runtime.h>

#define S_LEN   2048
#define D_MODEL 1024
#define NHEADS  16
#define HDIM    64
#define BATCH   4
#define WIN     128

typedef __bf16 v16bf __attribute__((ext_vector_type(16)));
typedef __bf16 v8bf  __attribute__((ext_vector_type(8)));
typedef float  v8f   __attribute__((ext_vector_type(8)));

static __device__ inline unsigned short f2bf(float f) {
    unsigned int u = __builtin_bit_cast(unsigned int, f);
    unsigned int r = (u + 0x7FFFu + ((u >> 16) & 1u)) >> 16;  // RNE
    return (unsigned short)r;
}

static __device__ inline v16bf cat16(v8bf lo, v8bf hi) {
    v16bf r;
#pragma unroll
    for (int i = 0; i < 8; ++i) { r[i] = lo[i]; r[i + 8] = hi[i]; }
    return r;
}

// ---------------------------------------------------------------- cast fp32->bf16
__global__ void cast_bf16(const float* __restrict__ src,
                          unsigned short* __restrict__ dst, int n) {
    int i = blockIdx.x * blockDim.x + threadIdx.x;
    int stride = gridDim.x * blockDim.x;
    for (; i < n; i += stride) dst[i] = f2bf(src[i]);
}

// ---------------------------------------------------------------- QKV projection
// X[8192,1024] @ W[1024,1024] + bias.  blockIdx.z: 0=Q (scaled, [B,H,S,64]),
// 1=K (transposed, [B,H,64,S]), 2=V ([B,H,S,64]).  All outputs bf16.
// Block tile 256x128, 8 waves, 64x64 per wave (4x4 WMMA).
#define XS_STRIDE 40    // 80B rows: 16B aligned, bank-spread
#define WS_STRIDE 136   // 272B rows: 16B aligned

__global__ void __launch_bounds__(256) qkv_gemm(
    const unsigned short* __restrict__ Xb,
    const unsigned short* __restrict__ Wqb,
    const unsigned short* __restrict__ Wkb,
    const unsigned short* __restrict__ Wvb,
    const float* __restrict__ bq, const float* __restrict__ bk,
    const float* __restrict__ bv,
    unsigned short* __restrict__ Qo, unsigned short* __restrict__ Kto,
    unsigned short* __restrict__ Vo)
{
    __shared__ unsigned short Xs[256 * XS_STRIDE];
    __shared__ unsigned short Ws[32 * WS_STRIDE];

    const int mode = blockIdx.z;
    const unsigned short* W = (mode == 0) ? Wqb : ((mode == 1) ? Wkb : Wvb);
    const float* bias       = (mode == 0) ? bq  : ((mode == 1) ? bk  : bv);

    const int t    = threadIdx.x;
    const int wave = t >> 5;
    const int lane = t & 31;
    const int lr   = lane & 15;
    const int hi   = lane >> 4;
    const int wm   = wave & 3;   // 4 waves along M (4*64 = 256)
    const int wn   = wave >> 2;  // 2 waves along N (2*64 = 128)

    const int m_blk = blockIdx.x * 256;
    const int n_blk = blockIdx.y * 128;

    v8f acc[4][4] = {};

    for (int k0 = 0; k0 < D_MODEL; k0 += 32) {
        // stage X tile 256x32 (bf16): 1024 chunks of 8
#pragma unroll
        for (int i = 0; i < 4; ++i) {
            int c   = t + i * 256;
            int row = c >> 2;
            int col = (c & 3) * 8;
            *(v8bf*)&Xs[row * XS_STRIDE + col] =
                *(const v8bf*)(Xb + (size_t)(m_blk + row) * D_MODEL + k0 + col);
        }
        // stage W tile 32x128: 512 chunks of 8
#pragma unroll
        for (int i = 0; i < 2; ++i) {
            int c   = t + i * 256;
            int row = c >> 4;
            int col = (c & 15) * 8;
            *(v8bf*)&Ws[row * WS_STRIDE + col] =
                *(const v8bf*)(W + (size_t)(k0 + row) * D_MODEL + n_blk + col);
        }
        __syncthreads();

        v16bf afr[4], bfr[4];
#pragma unroll
        for (int i = 0; i < 4; ++i) {
            const unsigned short* p = &Xs[(wm * 64 + i * 16 + lr) * XS_STRIDE + hi * 8];
            afr[i] = cat16(*(const v8bf*)p, *(const v8bf*)(p + 16));
        }
#pragma unroll
        for (int j = 0; j < 4; ++j)
#pragma unroll
            for (int s = 0; s < 16; ++s) {
                int kr = (s & 7) + 8 * hi + 16 * (s >> 3);
                bfr[j][s] = __builtin_bit_cast(__bf16, Ws[kr * WS_STRIDE + wn * 64 + j * 16 + lr]);
            }

#pragma unroll
        for (int i = 0; i < 4; ++i)
#pragma unroll
            for (int j = 0; j < 4; ++j)
                acc[i][j] = __builtin_amdgcn_wmma_f32_16x16x32_bf16(
                    false, afr[i], false, bfr[j], (short)0, acc[i][j], false, false);
        __syncthreads();
    }

#pragma unroll
    for (int i = 0; i < 4; ++i)
#pragma unroll
        for (int j = 0; j < 4; ++j)
#pragma unroll
            for (int r = 0; r < 8; ++r) {
                int mg = m_blk + wm * 64 + i * 16 + r + 8 * hi;
                int ng = n_blk + wn * 64 + j * 16 + lr;
                float v = acc[i][j][r] + bias[ng];
                int bb = mg >> 11, s = mg & 2047;
                int h  = ng >> 6,  d = ng & 63;
                int bh = bb * NHEADS + h;
                if (mode == 0)
                    Qo[((size_t)bh * S_LEN + s) * HDIM + d] = f2bf(v * 0.125f);
                else if (mode == 1)
                    Kto[((size_t)bh * HDIM + d) * S_LEN + s] = f2bf(v);
                else
                    Vo[((size_t)bh * S_LEN + s) * HDIM + d] = f2bf(v);
            }
}

// ---------------------------------------------------------------- sliding-window attention
// One wave handles 32 queries (2 M-subtiles); K/V fragments shared across them.
__global__ void __launch_bounds__(128) swa_attn(
    const unsigned short* __restrict__ Qb,
    const unsigned short* __restrict__ Ktb,
    const unsigned short* __restrict__ Vb,
    float* __restrict__ out)
{
    __shared__ unsigned short Psh[4][32 * 32];   // per-wave P staging (32q x 32k)

    const int t    = threadIdx.x;
    const int wave = t >> 5;
    const int lane = t & 31;
    const int lr   = lane & 15;
    const int hi   = lane >> 4;

    const int q0 = (blockIdx.x * 4 + wave) * 32;
    const int h  = blockIdx.y;
    const int b  = blockIdx.z;
    const int bh = b * NHEADS + h;

    // Q A-fragments: 2 M-subtiles x 2 d-chunks, held in regs for the whole loop
    v16bf aq[2][2];
#pragma unroll
    for (int i = 0; i < 2; ++i)
#pragma unroll
        for (int dc = 0; dc < 2; ++dc) {
            const unsigned short* p =
                Qb + ((size_t)bh * S_LEN + q0 + 16 * i + lr) * HDIM + dc * 32 + hi * 8;
            aq[i][dc] = cat16(*(const v8bf*)p, *(const v8bf*)(p + 16));
        }

    const unsigned short* Ktbase = Ktb + (size_t)bh * HDIM * S_LEN;
    const unsigned short* Vbase  = Vb + (size_t)bh * S_LEN * HDIM;

    float m_i[2][8], l_i[2][8];
    v8f o[2][4] = {};
#pragma unroll
    for (int i = 0; i < 2; ++i)
#pragma unroll
        for (int r = 0; r < 8; ++r) { m_i[i][r] = -3.0e38f; l_i[i][r] = 0.0f; }

    const int ks = (q0 >= WIN) ? (q0 - WIN) : 0;
    const int ke = (q0 + 32 + WIN < S_LEN) ? (q0 + 32 + WIN) : S_LEN;

    for (int k0 = ks; k0 < ke; k0 += 32) {
        // --- S = Q * K^T : K fragments loaded once, used by both M-subtiles
        v8f sf[2][2] = {};
#pragma unroll
        for (int f = 0; f < 2; ++f) {
            int kc = k0 + 16 * f;
            if (kc > S_LEN - 16) kc = S_LEN - 16;   // clamp; junk slots masked
#pragma unroll
            for (int dc = 0; dc < 2; ++dc) {
                v16bf bk2;
#pragma unroll
                for (int s = 0; s < 16; ++s) {
                    int kr = (s & 7) + 8 * hi + 16 * (s >> 3);
                    bk2[s] = __builtin_bit_cast(__bf16, Ktbase[(size_t)(dc * 32 + kr) * S_LEN + kc + lr]);
                }
#pragma unroll
                for (int i = 0; i < 2; ++i)
                    sf[i][f] = __builtin_amdgcn_wmma_f32_16x16x32_bf16(
                        false, aq[i][dc], false, bk2, (short)0, sf[i][f], false, false);
            }
        }
        // --- band mask
#pragma unroll
        for (int i = 0; i < 2; ++i)
#pragma unroll
            for (int f = 0; f < 2; ++f)
#pragma unroll
                for (int r = 0; r < 8; ++r) {
                    int q = q0 + 16 * i + r + 8 * hi;
                    int k = k0 + 16 * f + lr;
                    int dd = q - k; if (dd < 0) dd = -dd;
                    if (dd > WIN || k >= S_LEN) sf[i][f][r] = -1.0e30f;
                }
        // --- online softmax (per-row stats uniform in each 16-lane half,
        //     matching (m = 16i + r + 8*hi) accumulator distribution)
        float alpha[2][8];
#pragma unroll
        for (int i = 0; i < 2; ++i)
#pragma unroll
            for (int r = 0; r < 8; ++r) {
                float mx = fmaxf(sf[i][0][r], sf[i][1][r]);
#pragma unroll
                for (int off = 8; off >= 1; off >>= 1)
                    mx = fmaxf(mx, __shfl_xor(mx, off, 32));
                float mnew = fmaxf(m_i[i][r], mx);
                alpha[i][r] = __expf(m_i[i][r] - mnew);
                float p0 = __expf(sf[i][0][r] - mnew);
                float p1 = __expf(sf[i][1][r] - mnew);
                sf[i][0][r] = p0; sf[i][1][r] = p1;
                float rs = p0 + p1;
#pragma unroll
                for (int off = 8; off >= 1; off >>= 1)
                    rs += __shfl_xor(rs, off, 32);
                l_i[i][r] = l_i[i][r] * alpha[i][r] + rs;
                m_i[i][r] = mnew;
            }
#pragma unroll
        for (int i = 0; i < 2; ++i)
#pragma unroll
            for (int dt = 0; dt < 4; ++dt)
#pragma unroll
                for (int r = 0; r < 8; ++r)
                    o[i][dt][r] *= alpha[i][r];

        // --- C-layout -> A-layout via wave-private LDS (bf16 32x32 row-major)
#pragma unroll
        for (int i = 0; i < 2; ++i)
#pragma unroll
            for (int f = 0; f < 2; ++f)
#pragma unroll
                for (int r = 0; r < 8; ++r)
                    Psh[wave][(16 * i + r + 8 * hi) * 32 + 16 * f + lr] = f2bf(sf[i][f][r]);
        v16bf ap[2];
#pragma unroll
        for (int i = 0; i < 2; ++i) {
            const unsigned short* p = &Psh[wave][(16 * i + lr) * 32 + hi * 8];
            ap[i] = cat16(*(const v8bf*)p, *(const v8bf*)(p + 16));
        }
        // --- O += P * V : V fragments loaded once, used by both M-subtiles
#pragma unroll
        for (int dt = 0; dt < 4; ++dt) {
            v16bf bv2;
#pragma unroll
            for (int s = 0; s < 16; ++s) {
                int kk = k0 + (s & 7) + 8 * hi + 16 * (s >> 3); if (kk > S_LEN - 1) kk = S_LEN - 1;
                bv2[s] = __builtin_bit_cast(__bf16, Vbase[(size_t)kk * HDIM + dt * 16 + lr]);
            }
#pragma unroll
            for (int i = 0; i < 2; ++i)
                o[i][dt] = __builtin_amdgcn_wmma_f32_16x16x32_bf16(
                    false, ap[i], false, bv2, (short)0, o[i][dt], false, false);
        }
    }

    // --- normalize + store fp32 [B,S,H,64]
#pragma unroll
    for (int i = 0; i < 2; ++i)
#pragma unroll
        for (int r = 0; r < 8; ++r) {
            float inv = 1.0f / l_i[i][r];
            int s = q0 + 16 * i + r + 8 * hi;
#pragma unroll
            for (int dt = 0; dt < 4; ++dt) {
                int d = dt * 16 + lr;
                out[(((size_t)b * S_LEN + s) * NHEADS + h) * HDIM + d] = o[i][dt][r] * inv;
            }
        }
}

// ---------------------------------------------------------------- launch
extern "C" void kernel_launch(void* const* d_in, const int* in_sizes, int n_in,
                              void* d_out, int out_size, void* d_ws, size_t ws_size,
                              hipStream_t stream) {
    (void)in_sizes; (void)n_in; (void)out_size; (void)ws_size;
    const float* X  = (const float*)d_in[0];
    const float* Wq = (const float*)d_in[1];
    const float* bq = (const float*)d_in[2];
    const float* Wk = (const float*)d_in[3];
    const float* bk = (const float*)d_in[4];
    const float* Wv = (const float*)d_in[5];
    const float* bv = (const float*)d_in[6];
    float* out = (float*)d_out;

    const size_t NX = (size_t)BATCH * S_LEN * D_MODEL;       // 8388608
    const size_t NW = (size_t)D_MODEL * D_MODEL;             // 1048576
    const size_t NQ = (size_t)BATCH * NHEADS * S_LEN * HDIM; // 8388608

    unsigned short* ws  = (unsigned short*)d_ws;
    unsigned short* Xb  = ws;
    unsigned short* Wqb = Xb + NX;
    unsigned short* Wkb = Wqb + NW;
    unsigned short* Wvb = Wkb + NW;
    unsigned short* Qb  = Wvb + NW;
    unsigned short* Ktb = Qb + NQ;
    unsigned short* Vb  = Ktb + NQ;

    cast_bf16<<<2048, 256, 0, stream>>>(X,  Xb,  (int)NX);
    cast_bf16<<<1024, 256, 0, stream>>>(Wq, Wqb, (int)NW);
    cast_bf16<<<1024, 256, 0, stream>>>(Wk, Wkb, (int)NW);
    cast_bf16<<<1024, 256, 0, stream>>>(Wv, Wvb, (int)NW);

    dim3 ggrid(32, 8, 3);    // M/256, N/128, {Q,K,V}
    qkv_gemm<<<ggrid, 256, 0, stream>>>(Xb, Wqb, Wkb, Wvb, bq, bk, bv, Qb, Ktb, Vb);

    dim3 agrid(S_LEN / 128, NHEADS, BATCH);  // 4 waves x 32 queries per block
    swa_attn<<<agrid, 128, 0, stream>>>(Qb, Ktb, Vb, out);
}
